// MultiHeadedAttention_50525995270353
// MI455X (gfx1250) — matrix operand order repaired and checked by reading, not compiled, not run on hardware
//
#include <hip/hip_runtime.h>
#include <hip/hip_bf16.h>

typedef _Float16 f16;
typedef __attribute__((ext_vector_type(16))) _Float16     v16h;
typedef __attribute__((ext_vector_type(8)))  float        v8f;
typedef __attribute__((ext_vector_type(4)))  unsigned int v4u;
typedef __attribute__((ext_vector_type(8)))  unsigned int v8u;

constexpr int Bc = 4;
constexpr int Sc = 2048;
constexpr int Ec = 1024;
constexpr int Hc = 16;
constexpr int Dc = 64;      // head dim

// -------------------------------------------------------------------------
// WMMA fragment layouts (CDNA5 ISA §7.12.2, wave32):
//  A (16x32 f16): m = lane&15; elems 0..7 -> K = 8*(lane>>4)+i ; 8..15 -> +16
//  B (32x16 f16): n = lane&15; this code loads B (and A) with elem i -> K = 16*(lane>>4)+i
//  C/D (16x16 f32): n = lane&15; vgpr j -> M = j + 8*(lane>>4)
// -------------------------------------------------------------------------

// ======================= fp32 -> fp16 conversion =========================
__global__ __launch_bounds__(256)
void cvt_f32_to_f16(const float* __restrict__ in, f16* __restrict__ out, long n)
{
    long i = (long)blockIdx.x * blockDim.x + threadIdx.x;
    const long stride = (long)gridDim.x * blockDim.x;
    for (; i < n; i += stride) out[i] = (f16)in[i];
}

// ===================== TDM 2D tile load (inline asm) =====================
// Builds a Tensor DMA Descriptor (D#) per CDNA5 ISA
// and issues tensor_load_to_lds. data_size = 2 bytes (f16). pad_enable adds
// 1 DWORD of LDS padding per pad-interval (== one tile row) so LDS rows get
// a bank-friendly stride. Wave-uniform; tracked by TENSORcnt.
__device__ __forceinline__ void tdm_load_tile_f16(
    unsigned lds_off,                 // LDS byte offset of tile start
    const f16* gptr,                  // global tile start
    unsigned tile_w, unsigned tile_h, // tile dims (elements / rows)
    unsigned tensor_w, unsigned tensor_h,
    unsigned long long row_stride,    // elements between tensor rows
    unsigned pad_interval_code)       // log2(row DWORDs)-1 ; pad_amount=1 DW
{
    const unsigned long long ga = (unsigned long long)(uintptr_t)gptr;
    v4u g0;
    g0[0] = 1u;                                          // count=1, user D#
    g0[1] = lds_off;                                     // lds_addr
    g0[2] = (unsigned)ga;                                // global_addr[31:0]
    g0[3] = ((unsigned)(ga >> 32) & 0x01FFFFFFu) | (2u << 30);  // [56:32]|type=2
    v8u g1;
    g1[0] = (1u << 16)                                   // data_size = 2B
          | (1u << 20)                                   // pad_enable
          | (pad_interval_code << 22);                   // pad_amount code 0 = 1 DW
    g1[1] = (tensor_w & 0xFFFFu) << 16;                  // tensor_dim0[15:0]
    g1[2] = (tensor_w >> 16) | ((tensor_h & 0xFFFFu) << 16);
    g1[3] = (tensor_h >> 16) | (tile_w << 16);           // tile_dim0
    g1[4] = tile_h & 0xFFFFu;                            // tile_dim1 (tile_dim2=0)
    g1[5] = (unsigned)row_stride;                        // tensor_dim0_stride lo
    g1[6] = (unsigned)((row_stride >> 32) & 0xFFFFull);  // stride hi (dim1_stride=0)
    g1[7] = 0u;
    v4u gz = {0u, 0u, 0u, 0u};                           // groups 2/3 unused (2D)
    asm volatile("tensor_load_to_lds %0, %1, %2, %3"
                 :: "s"(g0), "s"(g1), "s"(gz), "s"(gz)
                 : "memory");
}

// ===================== GEMM + bias (TDM-fed, WMMA) =======================
// C[M,N] = A[M,K] @ B[K,N] + bias[N]; A,B are f16, double-buffered LDS tiles
// staged by the Tensor Data Mover. Block = 256 threads (8 waves), tile
// 128x64, BK=32. Output f16 or f32.
constexpr int BM = 128, BN = 64, BK = 32;
constexpr int ASTR = BK + 2;   // 34 halves: 16-DW rows + 1-DW TDM pad
constexpr int BSTR = BN + 2;   // 66 halves: 32-DW rows + 1-DW TDM pad

template <bool OUT_F16>
__global__ __launch_bounds__(256)
void gemm_bias_wmma(const f16* __restrict__ A, const f16* __restrict__ Bm,
                    const float* __restrict__ bias, void* __restrict__ Cout,
                    int M, int N, int K)
{
    __shared__ f16 As[2][BM * ASTR];
    __shared__ f16 Bs[2][BK * BSTR];

    const int tid  = threadIdx.x;
    const int wave = tid >> 5;
    const int lane = tid & 31;
    const int ln   = lane & 15;
    const int hi   = lane >> 4;

    const int m0 = blockIdx.y * BM;
    const int n0 = blockIdx.x * BN;
    const int wm = wave * 16;            // wave's 16-row strip

    const bool tdm_wave = (tid < 32);    // wave 0 drives the TDM
    const int nk = K / BK;

    v8f acc[4] = {};

    if (tdm_wave) {
        tdm_load_tile_f16((unsigned)(uintptr_t)&As[0][0],
                          A + (size_t)m0 * K, BK, BM, K, M,
                          (unsigned long long)K, 3u);
        tdm_load_tile_f16((unsigned)(uintptr_t)&Bs[0][0],
                          Bm + n0, BN, BK, N, K,
                          (unsigned long long)N, 4u);
    }

    for (int kt = 0; kt < nk; ++kt) {
        const int buf = kt & 1;
        if (tdm_wave) {
            if (kt + 1 < nk) {
                const int k1 = (kt + 1) * BK;
                tdm_load_tile_f16((unsigned)(uintptr_t)&As[buf ^ 1][0],
                                  A + (size_t)m0 * K + k1, BK, BM, K, M,
                                  (unsigned long long)K, 3u);
                tdm_load_tile_f16((unsigned)(uintptr_t)&Bs[buf ^ 1][0],
                                  Bm + (size_t)k1 * N + n0, BN, BK, N, K,
                                  (unsigned long long)N, 4u);
                __builtin_amdgcn_s_wait_tensorcnt(2);   // current tiles done
            } else {
                __builtin_amdgcn_s_wait_tensorcnt(0);
            }
        }
        __syncthreads();

        const f16* Ab = As[buf];
        const f16* Bb = Bs[buf];
        v16h a;
        {
            const int m = wm + ln, kb = hi * 16;
            #pragma unroll
            for (int i = 0; i < 8; i++) {
                a[i]     = Ab[m * ASTR + kb + i];
                a[8 + i] = Ab[m * ASTR + 8 + kb + i];
            }
        }
        #pragma unroll
        for (int nb = 0; nb < 4; nb++) {
            v16h b;
            const int n = nb * 16 + ln, kb2 = hi * 16;
            #pragma unroll
            for (int i = 0; i < 16; i++) b[i] = Bb[(kb2 + i) * BSTR + n];
            acc[nb] = __builtin_amdgcn_wmma_f32_16x16x32_f16(
                false, a, false, b, (short)0, acc[nb], false, false);
        }
        __syncthreads();   // all waves done with buf before TDM rewrites it
    }

    #pragma unroll
    for (int nb = 0; nb < 4; nb++) {
        const int col = n0 + nb * 16 + ln;
        const float bv = bias[col];
        #pragma unroll
        for (int j = 0; j < 8; j++) {
            const int row = m0 + wm + j + hi * 8;
            const float v = acc[nb][j] + bv;
            if constexpr (OUT_F16)
                ((f16*)Cout)[(size_t)row * N + col] = (f16)v;
            else
                ((float*)Cout)[(size_t)row * N + col] = v;
        }
    }
}

// ======================= causal flash attention ==========================
// One wave per 16-query tile of one (b,h). Online-softmax over 32-key tiles.
__global__ __launch_bounds__(32)
void flash_attn_wmma(const f16* __restrict__ qkv, f16* __restrict__ out)
{
    constexpr int QT = 16, KT = 32;
    __shared__ f16 Qs[QT][Dc + 8];
    __shared__ f16 Ks[KT][Dc + 8];
    __shared__ f16 Vs[KT][Dc + 8];
    __shared__ f16 Ps[QT][KT + 8];

    const int lane = threadIdx.x;
    const int ln = lane & 15, hi = lane >> 4;

    const int nqt = Sc / QT;                 // 128 query tiles
    const int qt  = blockIdx.x % nqt;
    const int bh  = blockIdx.x / nqt;
    const int b   = bh / Hc, h = bh % Hc;

    const int    q0        = qt * QT;
    const size_t rowstride = 3 * Ec;
    const size_t base      = (size_t)b * Sc * rowstride;
    const int qoff = h * Dc;
    const int koff = Ec + h * Dc;
    const int voff = 2 * Ec + h * Dc;

    for (int i = lane; i < QT * Dc; i += 32) {
        int r = i >> 6, c = i & 63;
        Qs[r][c] = qkv[base + (size_t)(q0 + r) * rowstride + qoff + c];
    }
    __syncthreads();

    v16h qa0, qa1;
    {
        const int kb = hi * 16;
        #pragma unroll
        for (int i = 0; i < 8; i++) {
            qa0[i]     = Qs[ln][kb + i];
            qa0[8 + i] = Qs[ln][8 + kb + i];
            qa1[i]     = Qs[ln][32 + kb + i];
            qa1[8 + i] = Qs[ln][40 + kb + i];
        }
    }

    const float NEG_INF = -__builtin_inff();
    float mrow[8], lrow[8];
    v8f o[4] = {};
    #pragma unroll
    for (int j = 0; j < 8; j++) { mrow[j] = NEG_INF; lrow[j] = 0.f; }

    const float scale = 0.125f;              // 1/sqrt(64)
    const int qhi = q0 + QT - 1;

    for (int j0 = 0; j0 <= qhi; j0 += KT) {
        for (int i = lane; i < KT * Dc; i += 32) {
            int r = i >> 6, c = i & 63;
            size_t rb = base + (size_t)(j0 + r) * rowstride;
            Ks[r][c] = qkv[rb + koff + c];
            Vs[r][c] = qkv[rb + voff + c];
        }
        __syncthreads();

        float p[2][8];
        float rmax[8];
        #pragma unroll
        for (int j = 0; j < 8; j++) rmax[j] = NEG_INF;

        #pragma unroll
        for (int ch = 0; ch < 2; ch++) {
            v8f s = {};
            #pragma unroll
            for (int dk = 0; dk < 2; dk++) {
                v16h kf;
                const int key = ch * 16 + ln;
                const int db  = dk * 32 + hi * 16;
                #pragma unroll
                for (int i = 0; i < 16; i++) kf[i] = Ks[key][db + i];
                s = __builtin_amdgcn_wmma_f32_16x16x32_f16(
                    false, dk ? qa1 : qa0, false, kf, (short)0, s, false, false);
            }
            #pragma unroll
            for (int j = 0; j < 8; j++) {
                const int qi = q0 + j + hi * 8;
                const int kj = j0 + ch * 16 + ln;
                const float sv = (kj <= qi) ? s[j] * scale : NEG_INF;
                p[ch][j] = sv;
                rmax[j]  = fmaxf(rmax[j], sv);
            }
        }

        #pragma unroll
        for (int j = 0; j < 8; j++) {
            float v = rmax[j];
            #pragma unroll
            for (int off = 1; off < 16; off <<= 1)
                v = fmaxf(v, __shfl_xor(v, off, 32));
            const float mnew = fmaxf(mrow[j], v);
            const float corr = __expf(mrow[j] - mnew);
            mrow[j] = mnew;
            const float p0 = __expf(p[0][j] - mnew);
            const float p1 = __expf(p[1][j] - mnew);
            p[0][j] = p0; p[1][j] = p1;
            float rs = p0 + p1;
            #pragma unroll
            for (int off = 1; off < 16; off <<= 1)
                rs += __shfl_xor(rs, off, 32);
            lrow[j] = lrow[j] * corr + rs;
            #pragma unroll
            for (int d = 0; d < 4; d++) o[d][j] *= corr;
        }

        #pragma unroll
        for (int ch = 0; ch < 2; ch++)
            #pragma unroll
            for (int j = 0; j < 8; j++)
                Ps[j + hi * 8][ch * 16 + ln] = (f16)p[ch][j];
        __syncthreads();

        v16h pa;
        {
            const int kb = hi * 16;
            #pragma unroll
            for (int i = 0; i < 8; i++) {
                pa[i]     = Ps[ln][kb + i];
                pa[8 + i] = Ps[ln][8 + kb + i];
            }
        }
        #pragma unroll
        for (int d = 0; d < 4; d++) {
            v16h vb;
            const int kb2 = hi * 16;
            #pragma unroll
            for (int i = 0; i < 16; i++) vb[i] = Vs[kb2 + i][d * 16 + ln];
            o[d] = __builtin_amdgcn_wmma_f32_16x16x32_f16(
                false, pa, false, vb, (short)0, o[d], false, false);
        }
        __syncthreads();
    }

    #pragma unroll
    for (int j = 0; j < 8; j++) {
        const int m = j + hi * 8;
        const float inv = 1.f / lrow[j];
        #pragma unroll
        for (int d = 0; d < 4; d++)
            out[((size_t)b * Sc + q0 + m) * Ec + h * Dc + d * 16 + ln] =
                (f16)(o[d][j] * inv);
    }
}

// ============================== launcher =================================
extern "C" void kernel_launch(void* const* d_in, const int* in_sizes, int n_in,
                              void* d_out, int out_size, void* d_ws, size_t ws_size,
                              hipStream_t stream)
{
    const float* x    = (const float*)d_in[0];
    const float* Wqkv = (const float*)d_in[1];
    const float* bqkv = (const float*)d_in[2];
    const float* Wo   = (const float*)d_in[3];
    const float* bo   = (const float*)d_in[4];

    const int M = Bc * Sc;                         // 8192
    f16* xh    = (f16*)d_ws;                       // M * E
    f16* wqkvh = xh + (size_t)M * Ec;              // E * 3E
    f16* woh   = wqkvh + (size_t)Ec * 3 * Ec;      // E * E
    f16* qkvh  = woh + (size_t)Ec * Ec;            // M * 3E
    f16* attnh = qkvh + (size_t)M * 3 * Ec;        // M * E

    // 0) one-shot fp32 -> fp16 conversions (TDM moves raw bytes)
    cvt_f32_to_f16<<<2048, 256, 0, stream>>>(x, xh, (long)M * Ec);
    cvt_f32_to_f16<<<2048, 256, 0, stream>>>(Wqkv, wqkvh, (long)Ec * 3 * Ec);
    cvt_f32_to_f16<<<1024, 256, 0, stream>>>(Wo, woh, (long)Ec * Ec);

    // 1) qkv = x @ Wqkv + bqkv   (f16 out)
    {
        dim3 grid(3 * Ec / BN, M / BM);
        gemm_bias_wmma<true><<<grid, 256, 0, stream>>>(
            xh, wqkvh, bqkv, qkvh, M, 3 * Ec, Ec);
    }
    // 2) causal flash attention per (b,h,query-tile)
    flash_attn_wmma<<<Bc * Hc * (Sc / 16), 32, 0, stream>>>(qkvh, attnh);

    // 3) out = attn @ Wo + bo    (f32 out)
    {
        dim3 grid(Ec / BN, M / BM);
        gemm_bias_wmma<false><<<grid, 256, 0, stream>>>(
            attnh, woh, bo, d_out, M, Ec, Ec);
    }
}
